// VectorQuantizerEMA_54142357733979
// MI455X (gfx1250) — compile-verified, matrix operand order checked by reading, not run on hardware
//
#include <hip/hip_runtime.h>
#include <hip/hip_bf16.h>

// ---------------- problem constants ----------------
#define N_TOK   32768      // 16*2048 flattened rows of z
#define EDIM    256
#define NE      8192
#define BETA    0.25f
#define EPS     1e-5f
#define ZQ_CNT  (N_TOK * EDIM)          // 8388608
#define IDX_OFF ZQ_CNT
#define LOSS_OFF (ZQ_CNT + N_TOK)       // 8421376
#define EMB_OFF  (LOSS_OFF + 1)         // 8421377

typedef __attribute__((ext_vector_type(16))) __bf16   v16bf;
typedef __attribute__((ext_vector_type(8)))  float    v8f;
typedef __attribute__((ext_vector_type(4)))  unsigned u32x4;
typedef __attribute__((ext_vector_type(4)))  float    f32x4;

union BFrag { v16bf v; u32x4 q[2]; unsigned u[8]; };

__device__ __forceinline__ unsigned pack_bf16(float a, float b) {
    unsigned ua = __float_as_uint(a);
    unsigned ub = __float_as_uint(b);
    ua += 0x7FFFu + ((ua >> 16) & 1u);   // round-to-nearest-even
    ub += 0x7FFFu + ((ub >> 16) & 1u);
    return (ua >> 16) | (ub & 0xFFFF0000u);
}

// 64B contiguous async copy global -> LDS (CDNA5 async path, ASYNCcnt).
// ISA: the instruction immediate offset is applied to BOTH the LDS dest
// address and the global source address, so offset:16/32/48 walks a
// contiguous 64-byte block on both sides.
__device__ __forceinline__ void async_copy64(unsigned lds_addr,
                                             unsigned long long gaddr) {
    asm volatile(
        "global_load_async_to_lds_b128 %0, %1, off\n\t"
        "global_load_async_to_lds_b128 %0, %1, off offset:16\n\t"
        "global_load_async_to_lds_b128 %0, %1, off offset:32\n\t"
        "global_load_async_to_lds_b128 %0, %1, off offset:48"
        :: "v"(lds_addr), "v"(gaddr) : "memory");
}

__device__ __forceinline__ void wait_async0() {
    asm volatile("s_wait_asynccnt 0x0" ::: "memory");
}

// ---- kernel 1: codebook f32 -> bf16 table + ||e||^2 per code -------------
__global__ __launch_bounds__(256) void vq_prep(const float* __restrict__ emb,
                                               unsigned short* __restrict__ ebf16,
                                               float* __restrict__ enorm) {
    const int lane = threadIdx.x & 31;
    const int wave = threadIdx.x >> 5;
    const int row  = blockIdx.x * 8 + wave;       // 1024 blocks * 8 waves = 8192
    const float* e = emb + (size_t)row * EDIM + lane * 8;
    f32x4 f0 = ((const f32x4*)e)[0];
    f32x4 f1 = ((const f32x4*)e)[1];
    u32x4 p;
    p.x = pack_bf16(f0.x, f0.y); p.y = pack_bf16(f0.z, f0.w);
    p.z = pack_bf16(f1.x, f1.y); p.w = pack_bf16(f1.z, f1.w);
    *(u32x4*)(ebf16 + (size_t)row * EDIM + lane * 8) = p;
    float s = f0.x*f0.x + f0.y*f0.y + f0.z*f0.z + f0.w*f0.w
            + f1.x*f1.x + f1.y*f1.y + f1.z*f1.z + f1.w*f1.w;
    #pragma unroll
    for (int off = 16; off; off >>= 1) s += __shfl_xor(s, off, 32);
    if (lane == 0) enorm[row] = s;
}

// ---- kernel 2: zero EMA accumulators + loss slot -------------------------
__global__ void vq_zero(float* __restrict__ p, int n, float* __restrict__ loss) {
    int i = blockIdx.x * blockDim.x + threadIdx.x;
    if (i == 0) *loss = 0.0f;
    for (; i < n; i += gridDim.x * blockDim.x) p[i] = 0.0f;
}

// ---- kernel 3: fused bf16-WMMA distance GEMM + row argmin ----------------
// grid = 256 blocks (128 rows each), block = 256 threads = 8 waves.
// Each wave: 16 rows of z. Loop: 256 N-tiles of 32 codes, double-buffered
// async global->LDS staging, two independent WMMA accumulator chains.
__global__ __launch_bounds__(256) void vq_argmin(const float* __restrict__ z,
                                                 const unsigned short* __restrict__ ebf16,
                                                 const float* __restrict__ enorm,
                                                 int*   __restrict__ idx_out,
                                                 float* __restrict__ idx_out_f) {
    __shared__ __align__(16) unsigned short Bt[2][32 * EDIM];   // 2 x 16 KB B tiles

    const int tid  = threadIdx.x;
    const int lane = tid & 31;
    const int wave = tid >> 5;
    const int hi   = lane >> 4;       // 0: lanes 0-15, 1: lanes 16-31
    const int l15  = lane & 15;
    const int rowBase = blockIdx.x * 128 + wave * 16;

    // Preload A fragments: this wave's 16 rows x 256 dims as bf16, in the ISA
    // 16-bit A layout (per 32-K chunk: lanes 0-15 hold K {0..7,16..23},
    // lanes 16-31 hold K {8..15,24..31}).
    BFrag a[8];
    {
        const float* zr = z + (size_t)(rowBase + l15) * EDIM;
        #pragma unroll
        for (int c = 0; c < 8; ++c) {
            const int o0 = c * 32 + hi * 8;
            f32x4 f0 = *(const f32x4*)(zr + o0);
            f32x4 f1 = *(const f32x4*)(zr + o0 + 4);
            f32x4 f2 = *(const f32x4*)(zr + o0 + 16);
            f32x4 f3 = *(const f32x4*)(zr + o0 + 20);
            a[c].u[0] = pack_bf16(f0.x, f0.y); a[c].u[1] = pack_bf16(f0.z, f0.w);
            a[c].u[2] = pack_bf16(f1.x, f1.y); a[c].u[3] = pack_bf16(f1.z, f1.w);
            a[c].u[4] = pack_bf16(f2.x, f2.y); a[c].u[5] = pack_bf16(f2.z, f2.w);
            a[c].u[6] = pack_bf16(f3.x, f3.y); a[c].u[7] = pack_bf16(f3.z, f3.w);
        }
    }

    float mn[8];
    int   mi[8];
    #pragma unroll
    for (int r = 0; r < 8; ++r) { mn[r] = 3.4e38f; mi[r] = 0; }

    // Per-thread staging slice: 32 codes * 512 B = 16 KB per tile;
    // 256 threads * 64 B each. thread -> (code = tid>>3, 64B chunk = tid&7).
    const int scode = tid >> 3;
    const int spart = tid & 7;
    const unsigned long long gbase =
        (unsigned long long)(const void*)ebf16 +
        ((unsigned long long)scode * EDIM + (unsigned)spart * 32) * 2ull;
    const unsigned ldsA =
        (unsigned)(size_t)(void*)&Bt[0][scode * EDIM + spart * 32];
    const unsigned ldsB =
        (unsigned)(size_t)(void*)&Bt[1][scode * EDIM + spart * 32];

    // prologue: stage tile 0 into buffer 0
    async_copy64(ldsA, gbase);

    for (int nt = 0; nt < NE / 32; ++nt) {
        wait_async0();       // this wave's loads for tile nt complete
        __syncthreads();     // all waves' loads done; buf[(nt+1)&1] free

        if (nt + 1 < NE / 32) {
            const unsigned lds = ((nt + 1) & 1) ? ldsB : ldsA;
            const unsigned long long ga =
                gbase + (unsigned long long)(nt + 1) * 32 * EDIM * 2;
            async_copy64(lds, ga);
        }
        if (nt + 2 < NE / 32)
            __builtin_prefetch(ebf16 + (size_t)(nt + 2) * 32 * EDIM, 0, 3);

        const unsigned short* B = Bt[nt & 1];
        const float en0 = enorm[nt * 32 + l15];
        const float en1 = enorm[nt * 32 + 16 + l15];

        v8f acc0 = {};
        v8f acc1 = {};
        #pragma unroll
        for (int c = 0; c < 8; ++c) {
            BFrag b0, b1;
            b0.q[0] = *(const u32x4*)(B + l15 * EDIM + c * 32 + hi * 8);
            b0.q[1] = *(const u32x4*)(B + l15 * EDIM + c * 32 + 16 + hi * 8);
            b1.q[0] = *(const u32x4*)(B + (16 + l15) * EDIM + c * 32 + hi * 8);
            b1.q[1] = *(const u32x4*)(B + (16 + l15) * EDIM + c * 32 + 16 + hi * 8);
            // two independent accumulator chains -> WMMA ILP
            acc0 = __builtin_amdgcn_wmma_f32_16x16x32_bf16(
                       false, a[c].v, false, b0.v, (short)0, acc0, false, false);
            acc1 = __builtin_amdgcn_wmma_f32_16x16x32_bf16(
                       false, a[c].v, false, b1.v, (short)0, acc1, false, false);
        }

        const int n0 = nt * 32 + l15;
        const int n1 = n0 + 16;
        #pragma unroll
        for (int r = 0; r < 8; ++r) {
            // C layout: VGPR r -> M = r + 8*hi, lane -> N = l15 (per group)
            float d0 = en0 - 2.0f * acc0[r];
            float d1 = en1 - 2.0f * acc1[r];
            float dm = d0; int nm = n0;
            if (d1 < d0) { dm = d1; nm = n1; }
            if (dm < mn[r]) { mn[r] = dm; mi[r] = nm; }
        }
    }

    // reduce (min, argmin) across the 16 lanes that share each row
    #pragma unroll
    for (int r = 0; r < 8; ++r) {
        float m = mn[r];
        int   i = mi[r];
        #pragma unroll
        for (int off = 1; off < 16; off <<= 1) {
            float om = __shfl_xor(m, off, 32);
            int   oi = __shfl_xor(i, off, 32);
            if (om < m || (om == m && oi < i)) { m = om; i = oi; }
        }
        if (l15 == 0) {
            const int row = rowBase + r + hi * 8;
            idx_out[row]   = i;
            idx_out_f[row] = (float)i;
        }
    }
}

// ---- kernel 4: gather z_q, loss reduction, EMA scatter -------------------
__global__ __launch_bounds__(256) void vq_gather(const float* __restrict__ z,
                                                 const float* __restrict__ emb,
                                                 const int*   __restrict__ idx,
                                                 float* __restrict__ zq_out,
                                                 float* __restrict__ loss_out,
                                                 float* __restrict__ dw,
                                                 float* __restrict__ bins) {
    const int n = blockIdx.x;        // 32768 rows
    const int d = threadIdx.x;       // 256 dims
    const int k = idx[n];
    const float zv = z[(size_t)n * EDIM + d];
    const float ev = emb[(size_t)k * EDIM + d];
    zq_out[(size_t)n * EDIM + d] = ev;   // z + sg(z_q - z) == z_q
    atomicAdd(&dw[(size_t)k * EDIM + d], zv);
    if (d == 0) atomicAdd(&bins[k], 1.0f);

    __shared__ float red[256];
    const float diff = zv - ev;
    red[d] = diff * diff;
    __syncthreads();
    #pragma unroll
    for (int s = 128; s; s >>= 1) {
        if (d < s) red[d] += red[d + s];
        __syncthreads();
    }
    if (d == 0) atomicAdd(loss_out, red[0] * (BETA / (float)ZQ_CNT));
}

// ---- kernel 5: codebook EMA update ---------------------------------------
__global__ __launch_bounds__(256) void vq_update(const float* __restrict__ dw,
                                                 const float* __restrict__ bins,
                                                 float* __restrict__ out) {
    const int k = blockIdx.x;        // 8192 codes
    const int d = threadIdx.x;
    const float nsum = (float)N_TOK;                       // sum of bins == N exactly
    const float cluster = (bins[k] + EPS) / (nsum + (float)NE * EPS) * nsum;
    out[(size_t)k * EDIM + d] = dw[(size_t)k * EDIM + d] / cluster;
}

// --------------------------------------------------------------------------
extern "C" void kernel_launch(void* const* d_in, const int* in_sizes, int n_in,
                              void* d_out, int out_size, void* d_ws, size_t ws_size,
                              hipStream_t stream) {
    const float* z   = (const float*)d_in[0];   // [16,2048,256]
    const float* emb = (const float*)d_in[1];   // [8192,256]
    (void)in_sizes; (void)n_in; (void)out_size; (void)ws_size;

    // workspace layout
    char* ws = (char*)d_ws;
    unsigned short* ebf16 = (unsigned short*)ws;                         // 4 MB
    float* enorm = (float*)(ws + 4u * 1024 * 1024);                      // 32 KB
    int*   idxw  = (int*)  (ws + 4u * 1024 * 1024 + 32 * 1024);          // 128 KB
    float* dw    = (float*)(ws + 4u * 1024 * 1024 + 160 * 1024);         // 8 MB
    float* bins  = (float*)(ws + 12u * 1024 * 1024 + 160 * 1024);        // 32 KB (contiguous after dw)

    // output layout (floats, concatenated tuple)
    float* out    = (float*)d_out;
    float* zq     = out;             // 8388608
    float* idxf   = out + IDX_OFF;   // 32768
    float* loss   = out + LOSS_OFF;  // 1
    float* newemb = out + EMB_OFF;   // 2097152

    vq_prep  <<<NE / 8, 256, 0, stream>>>(emb, ebf16, enorm);
    vq_zero  <<<2048, 256, 0, stream>>>(dw, NE * EDIM + NE, loss);  // dw+bins contiguous
    vq_argmin<<<N_TOK / 128, 256, 0, stream>>>(z, ebf16, enorm, idxw, idxf);
    vq_gather<<<N_TOK, 256, 0, stream>>>(z, emb, idxw, zq, loss, dw, bins);
    vq_update<<<NE, 256, 0, stream>>>(dw, bins, newemb);
}
